// Distill_25915832664372
// MI455X (gfx1250) — matrix operand order repaired and checked by reading, not compiled, not run on hardware
//
#include <hip/hip_runtime.h>

// ---------------------------------------------------------------------------
// Shapes: B=4, C=128, K=64, P=8, H=W=256 -> pp=64, L=1024, BQ = B*pp = 256.
// Patch q = (h/32)*8 + (w/32); l = (h%32)*32 + (w%32)  (contiguous 32x32 block)
// ---------------------------------------------------------------------------

typedef __attribute__((ext_vector_type(16))) __bf16 v16bf;
typedef __attribute__((ext_vector_type(8)))  float  v8f;

union FragBF {
  v16bf bf;
  uint4 q[2];
};

__device__ __forceinline__ unsigned short f2bf(float f) {
  union { float f; unsigned int u; } x; x.f = f;
  unsigned int u = x.u;
  u += 0x7fffu + ((u >> 16) & 1u);   // round-to-nearest-even
  return (unsigned short)(u >> 16);
}

__device__ __forceinline__ v8f vzero() {
  v8f z = {0.f, 0.f, 0.f, 0.f, 0.f, 0.f, 0.f, 0.f};
  return z;
}

__device__ __forceinline__ v8f wmma_bf16(const FragBF& a, const FragBF& b, v8f c) {
  return __builtin_amdgcn_wmma_f32_16x16x32_bf16(false, a.bf, false, b.bf,
                                                 (short)0, c, false, false);
}

// A fragment (16x32 bf16, M x K) from row-major [rows][ld] bf16 matrix.
__device__ __forceinline__ FragBF load_a(const unsigned short* p, int ld, int row0, int k0) {
  FragBF f;
  const int lane = threadIdx.x & 31;
  const int half = lane >> 4;
  const unsigned short* r = p + (size_t)(row0 + (lane & 15)) * ld + k0 + half * 16;
  f.q[0] = *(const uint4*)(r);
  f.q[1] = *(const uint4*)(r + 8);
  return f;
}

// B fragment (32x16 bf16, K x N) loaded from B^T stored row-major [N][ldk].
__device__ __forceinline__ FragBF load_bt(const unsigned short* pT, int ldk, int n0, int k0) {
  FragBF f;
  const int lane = threadIdx.x & 31;
  const int half = lane >> 4;
  const unsigned short* r = pT + (size_t)(n0 + (lane & 15)) * ldk + k0 + half * 16;
  f.q[0] = *(const uint4*)(r);
  f.q[1] = *(const uint4*)(r + 8);
  return f;
}

// B fragment from an LDS tile with 80-byte row pitch: row = n, 32 bytes at half*32.
__device__ __forceinline__ FragBF load_b_lds(const unsigned char* base, int row) {
  const int lane = threadIdx.x & 31;
  const int half = lane >> 4;
  const unsigned char* r = base + (size_t)(row + (lane & 15)) * 80 + half * 32;
  FragBF f;
  f.q[0] = *(const uint4*)(r);
  f.q[1] = *(const uint4*)(r + 16);
  return f;
}

// ---------------------------------------------------------------------------
// Kernel: f32 -> bf16 weight conversion
// ---------------------------------------------------------------------------
__global__ void k_f2bf(const float* __restrict__ src, unsigned short* __restrict__ dst, int n) {
  int i = blockIdx.x * blockDim.x + threadIdx.x;
  if (i < n) dst[i] = f2bf(src[i]);
}

// ---------------------------------------------------------------------------
// Kernel: img2patch + transpose + bf16 convert via LDS tile:
//   x[b,c,h,w] f32 -> Xt[bq][l][c] bf16, coalesced on both sides.
// grid = (BQ=256, 16 l-chunks of 64); block = 256
// ---------------------------------------------------------------------------
__global__ void k_transpose(const float* __restrict__ x1, const float* __restrict__ x2,
                            unsigned short* __restrict__ X1t, unsigned short* __restrict__ X2t) {
  const int bq = blockIdx.x;
  const int l0 = blockIdx.y * 64;
  const int b = bq >> 6, qq = bq & 63;
  const int hbase = ((qq >> 3) << 5) + ((l0 >> 5) & 31);  // l0 multiple of 64 -> (l0>>5) rows
  const int wbase = (qq & 7) << 5;

  __shared__ __align__(16) unsigned short Tsh[64][136];  // [l][c] padded

  const float* srcs[2] = {x1, x2};
  unsigned short* dsts[2] = {X1t, X2t};

  for (int img = 0; img < 2; ++img) {
    const float* x = srcs[img];
    // load: e = c*64 + ll ; consecutive tid -> consecutive w (coalesced 128B)
#pragma unroll
    for (int i = 0; i < 32; ++i) {
      int e = i * 256 + threadIdx.x;
      int c = e >> 6;
      int ll = e & 63;
      int hh = hbase + (ll >> 5);
      int ww = wbase + (ll & 31);
      float v = x[(((size_t)b * 128 + c) * 256 + hh) * 256 + ww];
      Tsh[ll][c] = f2bf(v);
    }
    __syncthreads();
    // store: e = ll*16 + seg ; 16 threads write 256B contiguous per l-row
    unsigned short* Xt = dsts[img];
#pragma unroll
    for (int i = 0; i < 4; ++i) {
      int e = i * 256 + threadIdx.x;
      int ll = e >> 4;
      int seg = e & 15;
      uint4 v = *(const uint4*)(&Tsh[ll][seg * 8]);
      *(uint4*)(Xt + ((size_t)(bq * 1024 + l0 + ll)) * 128 + seg * 8) = v;
    }
    __syncthreads();
  }
}

// ---------------------------------------------------------------------------
// Kernel: 1x1 conv GEMM  Y[o,l] = sum_c W[o,c]*X[c,l] + bias[o]
// B tile (128 l x 32 c) staged once per block via async global->LDS copy,
// double-buffered; all 8 waves share it.
// grid = (BQ=256, 8 l-tiles of 128); block = 256 (8 waves)
// MODE 0: store bf16 [bq][COUT][1024];  MODE 1: store f32 [bq][1024][COUT]
// ---------------------------------------------------------------------------
template <int COUT, int CIN, int MODE>
__global__ void k_conv(const unsigned short* __restrict__ Wb,  // [COUT][CIN] bf16
                       const float* __restrict__ bias,         // [COUT]
                       const unsigned short* __restrict__ Xt1, // [BQ][1024][128] bf16
                       const unsigned short* __restrict__ Xt2, // or nullptr
                       void* __restrict__ Yout) {
  constexpr int STRIPS = COUT / 16;     // 8 or 4
  constexpr int WPS    = 8 / STRIPS;    // waves per strip
  constexpr int COLS   = 128 / WPS;     // l-columns per wave
  constexpr int NT     = COLS / 16;     // output tiles per wave
  constexpr int NSTEP  = CIN / 32;      // K steps

  const int bq   = blockIdx.x;
  const int wave = threadIdx.x >> 5;
  const int lane = threadIdx.x & 31;
  const int strip = wave % STRIPS;
  const int lwoff = (wave / STRIPS) * COLS;      // wave's l offset within block tile
  const int l0blk = blockIdx.y * 128;

  const unsigned short* x1b = Xt1 + (size_t)bq * 1024 * 128;
  const unsigned short* x2b = (CIN == 256) ? (Xt2 + (size_t)bq * 1024 * 128) : Xt1;

  __shared__ __align__(16) unsigned char Bsh[2][128 * 80];  // 2 x 10KB, 80B row pitch
  const unsigned int ldsBase = (unsigned int)(unsigned long long)(&Bsh[0][0]);

  // Stage one 128x32 bf16 B tile: 512 x 16B segments, 2 async ops per thread.
  auto stage = [&](int kk, int buf) {
    const unsigned short* xb = (CIN == 256 && kk >= 128) ? x2b : x1b;
    const int kc = (CIN == 256 && kk >= 128) ? (kk - 128) : kk;
#pragma unroll
    for (int i = 0; i < 2; ++i) {
      int seg = threadIdx.x + i * 256;
      int row = seg >> 2;
      int s4 = seg & 3;
      unsigned long long ga = (unsigned long long)(const void*)(xb + (size_t)(l0blk + row) * 128 + kc + s4 * 8);
      unsigned int la = ldsBase + (unsigned int)(buf * (128 * 80) + row * 80 + s4 * 16);
      asm volatile("global_load_async_to_lds_b128 %0, %1, off"
                   :: "v"(la), "v"(ga) : "memory");
    }
  };

  v8f acc[NT];
#pragma unroll
  for (int t = 0; t < NT; ++t) acc[t] = vzero();

  stage(0, 0);
  FragBF a_cur = load_a(Wb, CIN, strip * 16, 0);

#pragma unroll
  for (int kks = 0; kks < NSTEP; ++kks) {
    asm volatile("s_wait_asynccnt 0x0" ::: "memory");
    __syncthreads();                       // buf (kks&1) ready for all waves
    if (kks + 1 < NSTEP) stage((kks + 1) * 32, (kks + 1) & 1);
    const int kkn = (kks + 1 < NSTEP) ? (kks + 1) * 32 : 0;
    FragBF a_nxt = load_a(Wb, CIN, strip * 16, kkn);
    const unsigned char* bbase = &Bsh[kks & 1][0];
#pragma unroll
    for (int t = 0; t < NT; ++t) {
      FragBF b = load_b_lds(bbase, lwoff + t * 16);
      acc[t] = wmma_bf16(a_cur, b, acc[t]);
    }
    a_cur = a_nxt;
  }

  const int half = lane >> 4;
  const int n = lane & 15;
  float bv[8];
#pragma unroll
  for (int r = 0; r < 8; ++r) bv[r] = bias[strip * 16 + half * 8 + r];

  if (MODE == 0) {
    unsigned short* Y = (unsigned short*)Yout + (size_t)bq * COUT * 1024;
#pragma unroll
    for (int t = 0; t < NT; ++t) {
#pragma unroll
      for (int r = 0; r < 8; ++r) {
        int o = strip * 16 + half * 8 + r;
        Y[(size_t)o * 1024 + (l0blk + lwoff + t * 16 + n)] = f2bf(acc[t][r] + bv[r]);
      }
    }
  } else {
    float* Y = (float*)Yout + (size_t)bq * 1024 * COUT;
#pragma unroll
    for (int t = 0; t < NT; ++t) {
#pragma unroll
      for (int r = 0; r < 8; ++r) {
        int o = strip * 16 + half * 8 + r;
        Y[(size_t)(l0blk + lwoff + t * 16 + n) * COUT + o] = acc[t][r] + bv[r];
      }
    }
  }
}

// ---------------------------------------------------------------------------
// Kernel: softmax over k=64 (contiguous) per (bq,l) column; write bf16 [l][k]
// ---------------------------------------------------------------------------
__global__ void k_softmax(const float* __restrict__ Vtmp, unsigned short* __restrict__ Vbf) {
  int idx = blockIdx.x * blockDim.x + threadIdx.x;  // 256*1024 columns
  const float4* s4 = (const float4*)(Vtmp + (size_t)idx * 64);
  float v[64];
  float m = -3.4e38f;
#pragma unroll
  for (int i = 0; i < 16; ++i) {
    float4 q = s4[i];
    v[4 * i + 0] = q.x; v[4 * i + 1] = q.y; v[4 * i + 2] = q.z; v[4 * i + 3] = q.w;
  }
#pragma unroll
  for (int i = 0; i < 64; ++i) m = fmaxf(m, v[i]);
  float sum = 0.f;
#pragma unroll
  for (int i = 0; i < 64; ++i) { v[i] = __expf(v[i] - m); sum += v[i]; }
  float inv = 1.f / sum;
  unsigned short* d = Vbf + (size_t)idx * 64;
#pragma unroll
  for (int i = 0; i < 8; ++i) {
    unsigned int pk[4];
#pragma unroll
    for (int j = 0; j < 4; ++j) {
      unsigned short lo = f2bf(v[i * 8 + 2 * j] * inv);
      unsigned short hi = f2bf(v[i * 8 + 2 * j + 1] * inv);
      pk[j] = (unsigned int)lo | ((unsigned int)hi << 16);
    }
    uint4 o; o.x = pk[0]; o.y = pk[1]; o.z = pk[2]; o.w = pk[3];
    *(uint4*)(d + i * 8) = o;
  }
}

// ---------------------------------------------------------------------------
// Kernel: U = l2norm_c(A @ Z^T), store transposed Ut[bq][k][c] bf16
// grid = BQ, block = 256 (8 waves); register double-buffered fragments.
// ---------------------------------------------------------------------------
__global__ void k_u(const unsigned short* __restrict__ A,  // [BQ][128][1024] bf16
                    const unsigned short* __restrict__ Z,  // [BQ][64][1024] bf16
                    unsigned short* __restrict__ Ut) {     // [BQ][64][128] bf16
  const int bq = blockIdx.x;
  const int wave = threadIdx.x >> 5;
  const int lane = threadIdx.x & 31;
  const unsigned short* Ab = A + (size_t)bq * 128 * 1024;
  const unsigned short* Zb = Z + (size_t)bq * 64 * 1024;

  v8f acc[4];
#pragma unroll
  for (int t = 0; t < 4; ++t) acc[t] = vzero();

  FragBF a = load_a(Ab, 1024, wave * 16, 0);
  FragBF b[4];
#pragma unroll
  for (int t = 0; t < 4; ++t) b[t] = load_bt(Zb, 1024, t * 16, 0);

  for (int ll = 0; ll < 1024; ll += 32) {
    const int lln = (ll + 32 < 1024) ? ll + 32 : 0;
    FragBF an = load_a(Ab, 1024, wave * 16, lln);
    FragBF bn[4];
#pragma unroll
    for (int t = 0; t < 4; ++t) bn[t] = load_bt(Zb, 1024, t * 16, lln);
#pragma unroll
    for (int t = 0; t < 4; ++t) acc[t] = wmma_bf16(a, b[t], acc[t]);
    a = an;
#pragma unroll
    for (int t = 0; t < 4; ++t) b[t] = bn[t];
  }

  __shared__ float ssum[64];
  if (threadIdx.x < 64) ssum[threadIdx.x] = 0.f;
  __syncthreads();

  const int half = lane >> 4;
  const int n = lane & 15;
#pragma unroll
  for (int t = 0; t < 4; ++t) {
    float ps = 0.f;
#pragma unroll
    for (int r = 0; r < 8; ++r) ps += acc[t][r] * acc[t][r];
    atomicAdd(&ssum[t * 16 + n], ps);  // ds_add_f32
  }
  __syncthreads();

#pragma unroll
  for (int t = 0; t < 4; ++t) {
    int k = t * 16 + n;
    float inv = 1.f / (1e-6f + sqrtf(ssum[k]));
    unsigned int pk[4];
#pragma unroll
    for (int j = 0; j < 4; ++j) {
      unsigned short lo = f2bf(acc[t][2 * j] * inv);
      unsigned short hi = f2bf(acc[t][2 * j + 1] * inv);
      pk[j] = (unsigned int)lo | ((unsigned int)hi << 16);
    }
    uint4 outv; outv.x = pk[0]; outv.y = pk[1]; outv.z = pk[2]; outv.w = pk[3];
    *(uint4*)(Ut + ((size_t)(bq * 64 + k) * 128 + wave * 16 + half * 8)) = outv;
  }
}

// ---------------------------------------------------------------------------
// Kernel: T = Wout @ U (in LDS, bf16), then Y3 = T @ V, written straight to
// out[b,o,h,w] via patch2img mapping (output conv fused via associativity).
// grid = BQ, block = 256 (8 waves); B fragments double-buffered in registers.
// ---------------------------------------------------------------------------
__global__ void k_out(const unsigned short* __restrict__ Ut,  // [BQ][64][128] bf16
                      const unsigned short* __restrict__ Wo,  // [128][128] bf16
                      const unsigned short* __restrict__ Vbf, // [BQ][1024][64] bf16
                      float* __restrict__ out) {              // [4][128][256][256]
  const int bq = blockIdx.x;
  const int wave = threadIdx.x >> 5;
  const int lane = threadIdx.x & 31;
  const int half = lane >> 4;
  const int n = lane & 15;
  const unsigned short* Utb = Ut + (size_t)bq * 64 * 128;
  const unsigned short* Vb = Vbf + (size_t)bq * 1024 * 64;

  __shared__ __align__(16) unsigned short Tsh[128 * 64];

  // Phase 1: T[c,k] = sum_cc Wout[c,cc] * U[cc,k]   (B^T = Ut[k][cc])
  {
    v8f acc[4];
#pragma unroll
    for (int t = 0; t < 4; ++t) acc[t] = vzero();
#pragma unroll
    for (int cc = 0; cc < 128; cc += 32) {
      FragBF a = load_a(Wo, 128, wave * 16, cc);
#pragma unroll
      for (int t = 0; t < 4; ++t) {
        FragBF b = load_bt(Utb, 128, t * 16, cc);
        acc[t] = wmma_bf16(a, b, acc[t]);
      }
    }
#pragma unroll
    for (int t = 0; t < 4; ++t) {
#pragma unroll
      for (int r = 0; r < 8; ++r) {
        int c = wave * 16 + half * 8 + r;
        Tsh[c * 64 + t * 16 + n] = f2bf(acc[t][r]);
      }
    }
  }
  __syncthreads();

  // Phase 2: Y3[c,l] = sum_k T[c,k] * V[k,l]; A frags resident, B double-buffered.
  FragBF a0 = load_a(Tsh, 64, wave * 16, 0);
  FragBF a1 = load_a(Tsh, 64, wave * 16, 32);
  const int bb = bq >> 6, qq = bq & 63;
  const int hbase = (qq >> 3) * 32, wbase = (qq & 7) * 32;

  FragBF b0 = load_bt(Vb, 64, 0, 0);
  FragBF b1 = load_bt(Vb, 64, 0, 32);
  for (int lt = 0; lt < 64; ++lt) {
    const int ltn = (lt + 1 < 64) ? lt + 1 : 0;
    FragBF nb0 = load_bt(Vb, 64, ltn * 16, 0);
    FragBF nb1 = load_bt(Vb, 64, ltn * 16, 32);
    v8f acc = vzero();
    acc = wmma_bf16(a0, b0, acc);
    acc = wmma_bf16(a1, b1, acc);
    int l = lt * 16 + n;
    int hh = hbase + (l >> 5);
    int ww = wbase + (l & 31);
#pragma unroll
    for (int r = 0; r < 8; ++r) {
      int o = wave * 16 + half * 8 + r;
      out[(((size_t)bb * 128 + o) * 256 + hh) * 256 + ww] = acc[r];
    }
    b0 = nb0; b1 = nb1;
  }
}

// ---------------------------------------------------------------------------
// Host launch
// ---------------------------------------------------------------------------
extern "C" void kernel_launch(void* const* d_in, const int* in_sizes, int n_in,
                              void* d_out, int out_size, void* d_ws, size_t ws_size,
                              hipStream_t stream) {
  const float* x1     = (const float*)d_in[0];
  const float* x2     = (const float*)d_in[1];
  const float* w_v    = (const float*)d_in[2];
  const float* b_v    = (const float*)d_in[3];
  const float* w1_1   = (const float*)d_in[4];
  const float* b1_1   = (const float*)d_in[5];
  const float* w2_1   = (const float*)d_in[6];
  const float* b2_1   = (const float*)d_in[7];
  const float* w_out1 = (const float*)d_in[8];
  const float* w1_2   = (const float*)d_in[9];
  const float* b1_2   = (const float*)d_in[10];
  const float* w2_2   = (const float*)d_in[11];
  const float* b2_2   = (const float*)d_in[12];
  const float* w_out2 = (const float*)d_in[13];

  const size_t BQ = 256, L = 1024, C = 128, KK = 64;

  char* ws = (char*)d_ws;
  size_t off = 0;
  auto alloc = [&](size_t bytes) -> char* {
    char* p = ws + off;
    off += (bytes + 255) & ~(size_t)255;
    return p;
  };

  unsigned short* X1t = (unsigned short*)alloc(BQ * L * C * 2);
  unsigned short* X2t = (unsigned short*)alloc(BQ * L * C * 2);
  unsigned short* A1  = (unsigned short*)alloc(BQ * C * L * 2);
  unsigned short* A2  = (unsigned short*)alloc(BQ * C * L * 2);
  unsigned short* Z1  = (unsigned short*)alloc(BQ * KK * L * 2);
  unsigned short* Z2  = (unsigned short*)alloc(BQ * KK * L * 2);
  float*          Vt  = (float*)         alloc(BQ * L * KK * 4);
  unsigned short* Vbf = (unsigned short*)alloc(BQ * L * KK * 2);
  unsigned short* Ut1 = (unsigned short*)alloc(BQ * KK * C * 2);
  unsigned short* Ut2 = (unsigned short*)alloc(BQ * KK * C * 2);
  unsigned short* w11b = (unsigned short*)alloc(128 * 128 * 2);
  unsigned short* w21b = (unsigned short*)alloc(64 * 128 * 2);
  unsigned short* w12b = (unsigned short*)alloc(128 * 128 * 2);
  unsigned short* w22b = (unsigned short*)alloc(64 * 128 * 2);
  unsigned short* wvb  = (unsigned short*)alloc(64 * 256 * 2);
  unsigned short* wo1b = (unsigned short*)alloc(128 * 128 * 2);
  unsigned short* wo2b = (unsigned short*)alloc(128 * 128 * 2);

  k_f2bf<<<64, 256, 0, stream>>>(w1_1, w11b, 128 * 128);
  k_f2bf<<<32, 256, 0, stream>>>(w2_1, w21b, 64 * 128);
  k_f2bf<<<64, 256, 0, stream>>>(w1_2, w12b, 128 * 128);
  k_f2bf<<<32, 256, 0, stream>>>(w2_2, w22b, 64 * 128);
  k_f2bf<<<64, 256, 0, stream>>>(w_v, wvb, 64 * 256);
  k_f2bf<<<64, 256, 0, stream>>>(w_out1, wo1b, 128 * 128);
  k_f2bf<<<64, 256, 0, stream>>>(w_out2, wo2b, 128 * 128);

  dim3 gtr(256, 16);
  k_transpose<<<gtr, 256, 0, stream>>>(x1, x2, X1t, X2t);

  dim3 gconv(256, 8);
  k_conv<128, 128, 0><<<gconv, 256, 0, stream>>>(w11b, b1_1, X1t, nullptr, A1);
  k_conv<128, 128, 0><<<gconv, 256, 0, stream>>>(w12b, b1_2, X2t, nullptr, A2);
  k_conv< 64, 128, 0><<<gconv, 256, 0, stream>>>(w21b, b2_1, X1t, nullptr, Z1);
  k_conv< 64, 128, 0><<<gconv, 256, 0, stream>>>(w22b, b2_2, X2t, nullptr, Z2);
  k_conv< 64, 256, 1><<<gconv, 256, 0, stream>>>(wvb, b_v, X1t, X2t, Vt);

  k_softmax<<<1024, 256, 0, stream>>>(Vt, Vbf);

  k_u<<<256, 256, 0, stream>>>(A1, Z1, Ut1);
  k_u<<<256, 256, 0, stream>>>(A2, Z2, Ut2);

  float* out1 = (float*)d_out;
  float* out2 = out1 + (size_t)4 * 128 * 256 * 256;
  k_out<<<256, 256, 0, stream>>>(Ut1, wo1b, Vbf, out1);
  k_out<<<256, 256, 0, stream>>>(Ut2, wo2b, Vbf, out2);
}
